// GModel_40312563040987
// MI455X (gfx1250) — matrix operand order repaired and checked by reading, not compiled, not run on hardware
//
#include <hip/hip_runtime.h>
#include <hip/hip_bf16.h>

typedef __attribute__((ext_vector_type(16))) _Float16 v16h;
typedef __attribute__((ext_vector_type(8)))  float    v8f;
typedef __attribute__((ext_vector_type(8)))  float    float8;

#define K_DIM   32768
#define B_ROWS  2048
#define CLASSES 10
#define KB_TOT  (K_DIM / 32)      // 1024 k-blocks of 32
#define SPLITS  8                 // K-split blocks (grid.y)
#define WAVES   8                 // waves per block
#define TILES   (B_ROWS / 16)     // 128 M-tiles

__device__ __forceinline__ float theta_c() { return 0.78539816339744830962f; } // 2*pi/8

// ---------------------------------------------------------------------------
// Kernel 1: pack cos(c*w), sin(c*w) into f16 in this code's contiguous-k WMMA B-operand
// layout: element [kb][lane][j] = trig(K = kb*32 + (lane/16)*16 + j, N = lane%16)
// N >= CLASSES is zero padding. 32768 threads, one (kb,lane) pair each.
// ---------------------------------------------------------------------------
__global__ void gsim_pack_w(const float* __restrict__ w,
                            _Float16* __restrict__ wc,
                            _Float16* __restrict__ wsn) {
    const int t    = blockIdx.x * blockDim.x + threadIdx.x;   // 0..32767
    const int kb   = t >> 5;
    const int lane = t & 31;
    const int n    = lane & 15;
    const int kbase = kb * 32 + (lane >> 4) * 16;
    const float c = theta_c();
    #pragma unroll
    for (int j = 0; j < 16; ++j) {
        float cv = 0.0f, sv = 0.0f;
        if (n < CLASSES) {
            float tw = c * w[(size_t)n * K_DIM + kbase + j];
            cv = __cosf(tw);
            sv = __sinf(tw);
        }
        wc[(size_t)t * 16 + j]  = (_Float16)cv;
        wsn[(size_t)t * 16 + j] = (_Float16)sv;
    }
}

// ---------------------------------------------------------------------------
// Kernel 2: fused trig + WMMA GEMM.
// grid = (TILES, SPLITS), block = 256 (8 waves). Wave (split,wave) reduces a
// 512-wide K chunk for one 16-row tile; 8 accumulators LDS-reduced per block.
// ---------------------------------------------------------------------------
__global__ void __launch_bounds__(256)
gsim_main(const float* __restrict__ x,
          const _Float16* __restrict__ wc,
          const _Float16* __restrict__ wsn,
          float* __restrict__ partial) {
    const int lane  = threadIdx.x & 31;
    const int wave  = threadIdx.x >> 5;
    const int tile  = blockIdx.x;          // 0..127
    const int split = blockIdx.y;          // 0..SPLITS-1
    const int half  = lane >> 4;           // lane group 0/1
    const int row   = tile * 16 + (lane & 15);

    const int chunk = split * WAVES + wave;              // 0..63
    const int kb0   = chunk * (KB_TOT / (SPLITS * WAVES)); // 16 kb-blocks per wave
    const int kb1   = kb0 + (KB_TOT / (SPLITS * WAVES));

    const float c = theta_c();
    const float* __restrict__ xrow = x + (size_t)row * K_DIM;

    v8f acc = {};
    for (int kb = kb0; kb < kb1; ++kb) {
        const int k0 = kb * 32;
        // A-operand: 16-bit A 16x32 layout. Lane loads two contiguous 32B
        // spans of its row: K = k0+16*half+[0..8) and K = k0+16*half+8+[0..8)
        float8 xa = *(const float8*)(xrow + k0 + 16 * half);
        float8 xb = *(const float8*)(xrow + k0 + 16 * half + 8);
        v16h ac, as;
        #pragma unroll
        for (int j = 0; j < 8; ++j) {
            float t = c * xa[j];
            ac[j] = (_Float16)__cosf(t);
            as[j] = (_Float16)__sinf(t);
        }
        #pragma unroll
        for (int j = 0; j < 8; ++j) {
            float t = c * xb[j];
            ac[8 + j] = (_Float16)__cosf(t);
            as[8 + j] = (_Float16)__sinf(t);
        }
        // B-operand: pre-packed, per-lane contiguous 32B
        v16h bc = *(const v16h*)(wc  + ((size_t)kb * 32 + lane) * 16);
        v16h bs = *(const v16h*)(wsn + ((size_t)kb * 32 + lane) * 16);

        acc = __builtin_amdgcn_wmma_f32_16x16x32_f16(
                  false, ac, false, bc, (short)0, acc, false, false);
        acc = __builtin_amdgcn_wmma_f32_16x16x32_f16(
                  false, as, false, bs, (short)0, acc, false, false);
    }

    // Reduce the 8 wave accumulators through LDS.
    __shared__ float red[WAVES][256];
    #pragma unroll
    for (int r = 0; r < 8; ++r) red[wave][lane * 8 + r] = acc[r];
    __syncthreads();

    float s = 0.0f;
    #pragma unroll
    for (int w = 0; w < WAVES; ++w) s += red[w][threadIdx.x];

    // Map LDS slot -> (b_local, o) using the C/D layout:
    // slot = L*8 + r ; M = r + 8*(L/16) ; N = L%16
    const int L = threadIdx.x >> 3;
    const int r = threadIdx.x & 7;
    const int b_local = r + 8 * (L >> 4);
    const int o = L & 15;
    partial[(((size_t)split * TILES + tile) * 16 + b_local) * 16 + o] = s;
}

// ---------------------------------------------------------------------------
// Kernel 3: sum the SPLITS partials, apply radius / sqrt(K), emit (2048,10).
// ---------------------------------------------------------------------------
__global__ void gsim_reduce(const float* __restrict__ partial,
                            const float* __restrict__ radius,
                            float* __restrict__ out) {
    const int t = blockIdx.x * blockDim.x + threadIdx.x;   // 0..32767
    const int b = t >> 4;
    const int o = t & 15;
    float s = 0.0f;
    #pragma unroll
    for (int sp = 0; sp < SPLITS; ++sp)
        s += partial[(((size_t)sp * TILES + (b >> 4)) * 16 + (b & 15)) * 16 + o];
    if (o < CLASSES)
        out[(size_t)b * CLASSES + o] = s * radius[0] * 0.005524271728019903f; // K^-1/2
}

extern "C" void kernel_launch(void* const* d_in, const int* in_sizes, int n_in,
                              void* d_out, int out_size, void* d_ws, size_t ws_size,
                              hipStream_t stream) {
    const float* x      = (const float*)d_in[0];
    const float* w      = (const float*)d_in[1];
    const float* radius = (const float*)d_in[2];
    float* out = (float*)d_out;

    char* wsb = (char*)d_ws;
    _Float16* wc      = (_Float16*)(wsb);                 // 1 MB
    _Float16* wsn     = (_Float16*)(wsb + (1u << 20));    // 1 MB
    float*    partial = (float*)   (wsb + (2u << 20));    // 1 MB

    gsim_pack_w<<<KB_TOT * 32 / 256, 256, 0, stream>>>(w, wc, wsn);
    gsim_main<<<dim3(TILES, SPLITS), 256, 0, stream>>>(x, wc, wsn, partial);
    gsim_reduce<<<B_ROWS * 16 / 256, 256, 0, stream>>>(partial, radius, out);
}
